// GNNImpute_9294309228794
// MI455X (gfx1250) — matrix operand order repaired and checked by reading, not compiled, not run on hardware
//
#include <hip/hip_runtime.h>

// ---------------------------------------------------------------------------
// Types for CDNA5 WMMA (gfx1250, wave32)
// ---------------------------------------------------------------------------
typedef __attribute__((ext_vector_type(16))) __bf16 v16bf;
typedef __attribute__((ext_vector_type(8)))  __bf16 v8bf;
typedef __attribute__((ext_vector_type(8)))  float  v8f;

#define BM 128
#define BN 64
#define BK 32

// ---------------------------------------------------------------------------
// GEMM: C[M,N] = A[M,K] x B[K,N] (+bias[N]), A/B bf16 in global, fp32 accum.
// Block = 256 threads = 8 waves; macro-tile 128x64; each wave owns 32x32
// (2 A-frags x 2 B-frags -> 4 WMMAs per K-step).
// Software-pipelined with LDS double buffering; the A tile is staged with
// CDNA5 async global->LDS copies (ASYNCcnt) overlapped with WMMA compute.
// ---------------------------------------------------------------------------
__global__ __launch_bounds__(256) void gemm_bf16_wmma(
    const __bf16* __restrict__ A, const __bf16* __restrict__ B,
    const float* __restrict__ bias, float* __restrict__ C,
    int M, int N, int K)
{
    __shared__ __align__(64) __bf16 As[2][BM][BK];   // 2 x 8 KB
    __shared__ __align__(64) __bf16 Bs[2][BN][BK];   // 2 x 4 KB (transposed [n][k])

    const int tid  = threadIdx.x;
    const int wave = tid >> 5;
    const int lane = tid & 31;
    const int half = lane >> 4;     // lane group 0/1
    const int l16  = lane & 15;

    const int blockM = blockIdx.y * BM;
    const int blockN = blockIdx.x * BN;
    const int wr = wave & 3;        // M sub-tile (32 rows each)
    const int wc = wave >> 2;       // N sub-tile (32 cols each)

    const bool interior = (blockM + BM <= M) && (blockN + BN <= N);
    const bool k4ok = ((K & 3) == 0);
    const int  nk = (K + BK - 1) / BK;

    v8f acc00 = {}, acc01 = {}, acc10 = {}, acc11 = {};

    auto stage = [&](int kk, int buf) {
        const bool full = interior && (kk + BK <= K);
        if (full) {
            // ---- A tile via CDNA5 async global->LDS copy ----
            if (k4ok) {
                #pragma unroll
                for (int j = 0; j < 4; ++j) {             // 128x32 bf16 = 1024 b64
                    int e = j * 256 + tid;
                    int r = e >> 3, c4 = (e & 7) * 4;
                    const __bf16* gp = A + (size_t)(blockM + r) * K + kk + c4;
                    unsigned lo = (unsigned)(size_t)&As[buf][r][c4];
                    asm volatile("global_load_async_to_lds_b64 %0, %1, off"
                                 :: "v"(lo), "v"(gp) : "memory");
                }
            } else {
                #pragma unroll
                for (int j = 0; j < 8; ++j) {             // 2048 b32
                    int e = j * 256 + tid;
                    int r = e >> 4, c2 = (e & 15) * 2;
                    const __bf16* gp = A + (size_t)(blockM + r) * K + kk + c2;
                    unsigned lo = (unsigned)(size_t)&As[buf][r][c2];
                    asm volatile("global_load_async_to_lds_b32 %0, %1, off"
                                 :: "v"(lo), "v"(gp) : "memory");
                }
            }
            // ---- B tile: register-staged transpose into LDS [n][k] ----
            #pragma unroll
            for (int j = 0; j < 4; ++j) {                 // 32x64 bf16 = 1024 b32
                int e = j * 256 + tid;
                int r = e >> 5, c2 = (e & 31) * 2;        // r = k row, c2 = n col
                unsigned pair = *(const unsigned*)(B + (size_t)(kk + r) * N + blockN + c2);
                __bf16 b0 = ((const __bf16*)&pair)[0];
                __bf16 b1 = ((const __bf16*)&pair)[1];
                Bs[buf][c2][r]     = b0;
                Bs[buf][c2 + 1][r] = b1;
            }
        } else {
            // ---- guarded tail: two-phase (batched loads, then LDS stores) ----
            __bf16 va[16];
            #pragma unroll
            for (int j = 0; j < 16; ++j) {                // 128x32 elements
                int e = j * 256 + tid;
                int r = e >> 5, c = e & 31;
                int gr = blockM + r, gc = kk + c;
                bool ok = (gr < M) && (gc < K);
                size_t idx = ok ? ((size_t)gr * K + gc) : 0;
                __bf16 v = A[idx];
                va[j] = ok ? v : (__bf16)0.0f;
            }
            #pragma unroll
            for (int j = 0; j < 16; ++j) {
                int e = j * 256 + tid;
                As[buf][e >> 5][e & 31] = va[j];
            }
            __bf16 vb[8];
            #pragma unroll
            for (int j = 0; j < 8; ++j) {                 // 32x64 elements
                int e = j * 256 + tid;
                int r = e >> 6, c = e & 63;
                int gk = kk + r, gn = blockN + c;
                bool ok = (gk < K) && (gn < N);
                size_t idx = ok ? ((size_t)gk * N + gn) : 0;
                __bf16 v = B[idx];
                vb[j] = ok ? v : (__bf16)0.0f;
            }
            #pragma unroll
            for (int j = 0; j < 8; ++j) {
                int e = j * 256 + tid;
                Bs[buf][e & 63][e >> 6] = vb[j];
            }
        }
    };

    stage(0, 0);
    for (int t = 0; t < nk; ++t) {
        const int buf = t & 1;
        // wait own async copies from previous stage, then block-wide barrier
        asm volatile("s_wait_asynccnt 0x0" ::: "memory");
        __syncthreads();
        // prefetch next tile into the other buffer (overlaps with WMMAs below)
        if (t + 1 < nk) stage((t + 1) * BK, buf ^ 1);

        // ---- fragments (contiguous k = 16*half + s in A and B) ----
        const int ar0 = wr * 32 + l16;
        const int ar1 = ar0 + 16;
        v8bf a0lo = *(const v8bf*)&As[buf][ar0][half * 16];
        v8bf a0hi = *(const v8bf*)&As[buf][ar0][half * 16 + 8];
        v8bf a1lo = *(const v8bf*)&As[buf][ar1][half * 16];
        v8bf a1hi = *(const v8bf*)&As[buf][ar1][half * 16 + 8];
        v16bf af0, af1;
        #pragma unroll
        for (int i = 0; i < 8; ++i) {
            af0[i] = a0lo[i]; af0[8 + i] = a0hi[i];
            af1[i] = a1lo[i]; af1[8 + i] = a1hi[i];
        }
        const int bc0 = wc * 32 + l16;
        v16bf bf0 = *(const v16bf*)&Bs[buf][bc0][half * 16];
        v16bf bf1 = *(const v16bf*)&Bs[buf][bc0 + 16][half * 16];

        acc00 = __builtin_amdgcn_wmma_f32_16x16x32_bf16(false, af0, false, bf0, (short)0, acc00, false, false);
        acc01 = __builtin_amdgcn_wmma_f32_16x16x32_bf16(false, af0, false, bf1, (short)0, acc01, false, false);
        acc10 = __builtin_amdgcn_wmma_f32_16x16x32_bf16(false, af1, false, bf0, (short)0, acc10, false, false);
        acc11 = __builtin_amdgcn_wmma_f32_16x16x32_bf16(false, af1, false, bf1, (short)0, acc11, false, false);
    }

    // ---- store: VGPR r -> row (r + 8*half), col = lane&15 ----
    const int row0 = blockM + wr * 32 + half * 8;
    const int col0 = blockN + wc * 32 + l16;
    #pragma unroll
    for (int r = 0; r < 8; ++r) {
        int ra = row0 + r, rb = ra + 16;
        int ca = col0,     cb = col0 + 16;
        float biasA = (bias && ca < N) ? bias[ca] : 0.0f;
        float biasB = (bias && cb < N) ? bias[cb] : 0.0f;
        if (ra < M) {
            if (ca < N) C[(size_t)ra * N + ca] = acc00[r] + biasA;
            if (cb < N) C[(size_t)ra * N + cb] = acc01[r] + biasB;
        }
        if (rb < M) {
            if (ca < N) C[(size_t)rb * N + ca] = acc10[r] + biasA;
            if (cb < N) C[(size_t)rb * N + cb] = acc11[r] + biasB;
        }
    }
}

// ---------------------------------------------------------------------------
// fp32 -> bf16 conversion (x as bf16 = 120 MB fits the 192 MB L2, so GEMM
// column-block re-reads of A hit cache instead of HBM).
// ---------------------------------------------------------------------------
__global__ void cvt_f32_bf16(const float* __restrict__ in, __bf16* __restrict__ out, size_t n)
{
    size_t i = (size_t)blockIdx.x * blockDim.x + threadIdx.x;
    if (i < n) out[i] = (__bf16)in[i];
}

// ---------------------------------------------------------------------------
// Attention scores: es[n,h] = <h[n,h,:], a_src[h,:]>, ed likewise.
// Wave32 per node; head = blockIdx.y (no integer division anywhere).
// ---------------------------------------------------------------------------
__global__ __launch_bounds__(256) void attn_scores(
    const float* __restrict__ hmat, const float* __restrict__ a_src,
    const float* __restrict__ a_dst, float* __restrict__ es,
    float* __restrict__ ed, int n, int Hh, int C)
{
    int node = blockIdx.x * 8 + (threadIdx.x >> 5);
    int hh   = blockIdx.y;
    int lane = threadIdx.x & 31;
    if (node >= n) return;
    const float* hp = hmat + (size_t)node * Hh * C + (size_t)hh * C;
    const float* as = a_src + (size_t)hh * C;
    const float* ad = a_dst + (size_t)hh * C;
    float s0 = 0.0f, s1 = 0.0f;
    for (int c = lane; c < C; c += 32) {
        float hv = hp[c];
        s0 += hv * as[c];
        s1 += hv * ad[c];
    }
    #pragma unroll
    for (int off = 16; off; off >>= 1) {
        s0 += __shfl_xor(s0, off, 32);
        s1 += __shfl_xor(s1, off, 32);
    }
    if (lane == 0) {
        es[(size_t)node * Hh + hh] = s0;
        ed[(size_t)node * Hh + hh] = s1;
    }
}

// ---------------------------------------------------------------------------
// Segment softmax helpers
// ---------------------------------------------------------------------------
__device__ inline void atomicMaxF(float* addr, float v)
{
    if (v >= 0.0f) atomicMax((int*)addr, __float_as_int(v));
    else           atomicMin((unsigned int*)addr, __float_as_uint(v));
}

__global__ void fill_kernel(float* __restrict__ p, float v, size_t n)
{
    size_t i = (size_t)blockIdx.x * blockDim.x + threadIdx.x;
    if (i < n) p[i] = v;
}

// one thread per edge, small loop over heads
__global__ void edge_logits_max(
    const long long* __restrict__ ei, int E, int Nn, int Hh,
    const float* __restrict__ es, const float* __restrict__ ed,
    float* __restrict__ el, float* __restrict__ m)
{
    int e = blockIdx.x * blockDim.x + threadIdx.x;
    if (e >= E + Nn) return;
    long long s, d;
    if (e < E) { s = ei[e]; d = ei[(size_t)E + e]; } else { s = d = e - E; }
    for (int hh = 0; hh < Hh; ++hh) {
        float v = es[(size_t)s * Hh + hh] + ed[(size_t)d * Hh + hh];
        v = (v > 0.0f) ? v : 0.2f * v;              // leaky_relu 0.2
        el[(size_t)e * Hh + hh] = v;
        atomicMaxF(&m[(size_t)d * Hh + hh], v);
    }
}

__global__ void edge_exp_sum(
    const long long* __restrict__ ei, int E, int Nn, int Hh,
    float* __restrict__ el, const float* __restrict__ m,
    float* __restrict__ denom)
{
    int e = blockIdx.x * blockDim.x + threadIdx.x;
    if (e >= E + Nn) return;
    long long d;
    if (e < E) d = ei[(size_t)E + e]; else d = e - E;
    for (int hh = 0; hh < Hh; ++hh) {
        float ex = __expf(el[(size_t)e * Hh + hh] - m[(size_t)d * Hh + hh]);
        el[(size_t)e * Hh + hh] = ex;
        atomicAdd(&denom[(size_t)d * Hh + hh], ex);
    }
}

// one block per (edge, head); edge indices are block-uniform (scalar loads)
__global__ void scatter_msg(
    const long long* __restrict__ ei, int E, int Nn, int Hh, int C,
    const float* __restrict__ hsrc, const float* __restrict__ ex,
    const float* __restrict__ denom, float* __restrict__ agg)
{
    int e  = blockIdx.x;
    int hh = blockIdx.y;
    long long s, d;
    if (e < E) { s = ei[e]; d = ei[(size_t)E + e]; } else { s = d = e - E; }
    float alpha = ex[(size_t)e * Hh + hh] / (denom[(size_t)d * Hh + hh] + 1e-16f);
    const float* hp = hsrc + (size_t)s * Hh * C + (size_t)hh * C;
    float*       ap = agg  + (size_t)d * Hh * C + (size_t)hh * C;
    for (int c = threadIdx.x; c < C; c += blockDim.x)
        atomicAdd(&ap[c], hp[c] * alpha);
}

// out[n,c] = mean_h agg[n,h,c] + b[c] ; node = blockIdx.y
__global__ void mean_heads_bias(
    const float* __restrict__ agg, const float* __restrict__ b,
    float* __restrict__ out, int Hh, int C)
{
    int c = blockIdx.x * blockDim.x + threadIdx.x;
    int node = blockIdx.y;
    if (c >= C) return;
    float s = 0.0f;
    for (int h = 0; h < Hh; ++h) s += agg[(size_t)node * Hh * C + (size_t)h * C + c];
    out[(size_t)node * C + c] = s / (float)Hh + b[c];
}

__global__ __launch_bounds__(256) void bn_stats(
    const float* __restrict__ x, float* __restrict__ mu,
    float* __restrict__ var, int n, int C)
{
    int c = blockIdx.x;
    float s = 0.0f, s2 = 0.0f;
    for (int r = threadIdx.x; r < n; r += 256) {
        float v = x[(size_t)r * C + c];
        s += v; s2 += v * v;
    }
    __shared__ float ss[256], ss2[256];
    ss[threadIdx.x] = s; ss2[threadIdx.x] = s2;
    __syncthreads();
    for (int off = 128; off; off >>= 1) {
        if (threadIdx.x < off) {
            ss[threadIdx.x]  += ss[threadIdx.x + off];
            ss2[threadIdx.x] += ss2[threadIdx.x + off];
        }
        __syncthreads();
    }
    if (threadIdx.x == 0) {
        float mean = ss[0] / (float)n;
        mu[c]  = mean;
        var[c] = ss2[0] / (float)n - mean * mean;
    }
}

// row = blockIdx.y
__global__ void bn_apply_relu(
    float* __restrict__ x, const float* __restrict__ mu,
    const float* __restrict__ var, const float* __restrict__ g,
    const float* __restrict__ be, int C)
{
    int c = blockIdx.x * blockDim.x + threadIdx.x;
    if (c >= C) return;
    size_t idx = (size_t)blockIdx.y * C + c;
    float v = (x[idx] - mu[c]) * rsqrtf(var[c] + 1e-5f) * g[c] + be[c];
    x[idx] = (v > 0.0f) ? v : 0.0f;
}

// row = blockIdx.y
__global__ void relu_scale(
    float* __restrict__ x, const float* __restrict__ sf, int C)
{
    int c = blockIdx.x * blockDim.x + threadIdx.x;
    if (c >= C) return;
    size_t idx = (size_t)blockIdx.y * C + c;
    float v = x[idx];
    x[idx] = ((v > 0.0f) ? v : 0.0f) * sf[blockIdx.y];
}

// ---------------------------------------------------------------------------
// Orchestration
// ---------------------------------------------------------------------------
static inline unsigned int cdiv_u(size_t a, size_t b) { return (unsigned int)((a + b - 1) / b); }

extern "C" void kernel_launch(void* const* d_in, const int* in_sizes, int n_in,
                              void* d_out, int out_size, void* d_ws, size_t ws_size,
                              hipStream_t stream)
{
    const float*     x   = (const float*)d_in[0];
    const long long* ei  = (const long long*)d_in[1];   // int64 edge_index [2,E]
    const float*     sf  = (const float*)d_in[2];
    const float*     W1  = (const float*)d_in[3];
    const float*     a1s = (const float*)d_in[4];
    const float*     a1d = (const float*)d_in[5];
    const float*     b1  = (const float*)d_in[6];
    const float*     g1  = (const float*)d_in[7];
    const float*     be1 = (const float*)d_in[8];
    const float*     W2  = (const float*)d_in[9];
    const float*     a2s = (const float*)d_in[10];
    const float*     a2d = (const float*)d_in[11];
    const float*     b2  = (const float*)d_in[12];
    const float*     g2  = (const float*)d_in[13];
    const float*     be2 = (const float*)d_in[14];
    const float*     dW1 = (const float*)d_in[15];
    const float*     db1 = (const float*)d_in[16];
    const float*     dg1 = (const float*)d_in[17];
    const float*     dbe1= (const float*)d_in[18];
    const float*     dW2 = (const float*)d_in[19];
    const float*     db2 = (const float*)d_in[20];
    float*           out = (float*)d_out;

    const int E    = in_sizes[1] / 2;
    const int N    = in_sizes[2];
    const int D    = in_sizes[0] / N;
    const int HD   = in_sizes[6];
    const int Hh   = in_sizes[4] / HD;
    const int ZD   = in_sizes[12];
    const int HC1  = Hh * HD;   // 1536
    const int HC2  = Hh * ZD;   // 150
    const int Etot = E + N;

    // ---------------- workspace carve-up (f32 units) ------------------------
    float* ws = (float*)d_ws;
    size_t off = 0;
    float* h1   = ws + off; off += (size_t)N * HC1;     // layer-1 features (f32)
    float* agg1 = ws + off; off += (size_t)N * HC1;
    float* hmid = ws + off; off += (size_t)N * HD;
    float* es   = ws + off; off += (size_t)N * Hh;
    float* ed   = ws + off; off += (size_t)N * Hh;
    float* mbuf = ws + off; off += (size_t)N * Hh;
    float* den  = ws + off; off += (size_t)N * Hh;
    float* el   = ws + off; off += (size_t)Etot * Hh;
    float* mu   = ws + off; off += 4096;
    float* var  = ws + off; off += 4096;

    // bf16 overlays into regions that are dead at time of use:
    __bf16* xb  = (__bf16*)agg1;        // used only in GEMM1 (before agg1 zero-fill)
    __bf16* W1b = (__bf16*)hmid;        // used only in GEMM1 (before hmid written)

    // layer-2 / decoder buffers reuse the h1 region once layer-1 scatter retires it
    float*  h2    = h1;                                   // N*HC2 f32
    float*  agg2  = h1 + (size_t)N * HC2;                 // N*HC2 f32
    float*  zbuf  = agg2 + (size_t)N * HC2;               // N*ZD  f32
    float*  dec1  = zbuf + (size_t)N * ZD;                // N*HD  f32
    float*  tail  = dec1 + (size_t)N * HD;
    __bf16* hmidb = (__bf16*)tail;                        // N*HD bf16
    __bf16* decb  = hmidb + (size_t)N * HD;               // N*HD bf16
    __bf16* zb    = decb + (size_t)N * HD;                // N*ZD bf16
    __bf16* W2b   = zb + (size_t)N * ZD;                  // HD*HC2 bf16
    __bf16* dW1b  = W2b + (size_t)HD * HC2;               // ZD*HD bf16
    __bf16* dW2b  = dW1b + (size_t)ZD * HD;               // HD*D bf16

    const int T = 256;
    const int scatT1 = (HD >= 256) ? 256 : 64;
    const int scatT2 = (ZD >= 256) ? 256 : 64;

    // ---------------- Encoder layer 1 (GAT: D -> HD, H heads) --------------
    cvt_f32_bf16<<<cdiv_u((size_t)N * D, T), T, 0, stream>>>(x, xb, (size_t)N * D);
    cvt_f32_bf16<<<cdiv_u((size_t)D * HC1, T), T, 0, stream>>>(W1, W1b, (size_t)D * HC1);
    {
        dim3 g(cdiv_u(HC1, BN), cdiv_u(N, BM));
        gemm_bf16_wmma<<<g, T, 0, stream>>>(xb, W1b, nullptr, h1, N, HC1, D);
    }
    attn_scores<<<dim3(cdiv_u(N, 8), Hh), T, 0, stream>>>(h1, a1s, a1d, es, ed, N, Hh, HD);
    fill_kernel<<<cdiv_u((size_t)N * Hh, T), T, 0, stream>>>(mbuf, -3.402823466e+38f, (size_t)N * Hh);
    fill_kernel<<<cdiv_u((size_t)N * Hh, T), T, 0, stream>>>(den, 0.0f, (size_t)N * Hh);
    edge_logits_max<<<cdiv_u(Etot, T), T, 0, stream>>>(ei, E, N, Hh, es, ed, el, mbuf);
    edge_exp_sum<<<cdiv_u(Etot, T), T, 0, stream>>>(ei, E, N, Hh, el, mbuf, den);
    fill_kernel<<<cdiv_u((size_t)N * HC1, T), T, 0, stream>>>(agg1, 0.0f, (size_t)N * HC1);
    scatter_msg<<<dim3(Etot, Hh), scatT1, 0, stream>>>(ei, E, N, Hh, HD, h1, el, den, agg1);
    mean_heads_bias<<<dim3(cdiv_u(HD, T), N), T, 0, stream>>>(agg1, b1, hmid, Hh, HD);
    bn_stats<<<HD, T, 0, stream>>>(hmid, mu, var, N, HD);
    bn_apply_relu<<<dim3(cdiv_u(HD, T), N), T, 0, stream>>>(hmid, mu, var, g1, be1, HD);

    // h1 region is now free: stage bf16 copies of layer-2 / decoder operands
    cvt_f32_bf16<<<cdiv_u((size_t)N * HD, T), T, 0, stream>>>(hmid, hmidb, (size_t)N * HD);
    cvt_f32_bf16<<<cdiv_u((size_t)HD * HC2, T), T, 0, stream>>>(W2, W2b, (size_t)HD * HC2);
    cvt_f32_bf16<<<cdiv_u((size_t)ZD * HD, T), T, 0, stream>>>(dW1, dW1b, (size_t)ZD * HD);
    cvt_f32_bf16<<<cdiv_u((size_t)HD * D, T), T, 0, stream>>>(dW2, dW2b, (size_t)HD * D);

    // ---------------- Encoder layer 2 (GAT: HD -> ZD, H heads) -------------
    {
        dim3 g(cdiv_u(HC2, BN), cdiv_u(N, BM));
        gemm_bf16_wmma<<<g, T, 0, stream>>>(hmidb, W2b, nullptr, h2, N, HC2, HD);
    }
    attn_scores<<<dim3(cdiv_u(N, 8), Hh), T, 0, stream>>>(h2, a2s, a2d, es, ed, N, Hh, ZD);
    fill_kernel<<<cdiv_u((size_t)N * Hh, T), T, 0, stream>>>(mbuf, -3.402823466e+38f, (size_t)N * Hh);
    fill_kernel<<<cdiv_u((size_t)N * Hh, T), T, 0, stream>>>(den, 0.0f, (size_t)N * Hh);
    edge_logits_max<<<cdiv_u(Etot, T), T, 0, stream>>>(ei, E, N, Hh, es, ed, el, mbuf);
    edge_exp_sum<<<cdiv_u(Etot, T), T, 0, stream>>>(ei, E, N, Hh, el, mbuf, den);
    fill_kernel<<<cdiv_u((size_t)N * HC2, T), T, 0, stream>>>(agg2, 0.0f, (size_t)N * HC2);
    scatter_msg<<<dim3(Etot, Hh), scatT2, 0, stream>>>(ei, E, N, Hh, ZD, h2, el, den, agg2);
    mean_heads_bias<<<dim3(cdiv_u(ZD, T), N), T, 0, stream>>>(agg2, b2, zbuf, Hh, ZD);
    bn_stats<<<ZD, T, 0, stream>>>(zbuf, mu, var, N, ZD);
    bn_apply_relu<<<dim3(cdiv_u(ZD, T), N), T, 0, stream>>>(zbuf, mu, var, g2, be2, ZD);
    cvt_f32_bf16<<<cdiv_u((size_t)N * ZD, T), T, 0, stream>>>(zbuf, zb, (size_t)N * ZD);

    // ---------------- Decoder ----------------------------------------------
    {
        dim3 g(cdiv_u(HD, BN), cdiv_u(N, BM));
        gemm_bf16_wmma<<<g, T, 0, stream>>>(zb, dW1b, db1, dec1, N, HD, ZD);
    }
    bn_stats<<<HD, T, 0, stream>>>(dec1, mu, var, N, HD);
    bn_apply_relu<<<dim3(cdiv_u(HD, T), N), T, 0, stream>>>(dec1, mu, var, dg1, dbe1, HD);
    cvt_f32_bf16<<<cdiv_u((size_t)N * HD, T), T, 0, stream>>>(dec1, decb, (size_t)N * HD);
    {
        dim3 g(cdiv_u(D, BN), cdiv_u(N, BM));
        gemm_bf16_wmma<<<g, T, 0, stream>>>(decb, dW2b, db2, out, N, D, HD);
    }
    relu_scale<<<dim3(cdiv_u(D, T), N), T, 0, stream>>>(out, sf, D);
}
